// Transformer_47090021433931
// MI455X (gfx1250) — matrix operand order repaired and checked by reading, not compiled, not run on hardware
//
#include <hip/hip_runtime.h>

// ---------------------------------------------------------------------------
// CDNA5 (gfx1250) transformer forward.  wave32, WMMA f32_16x16x32_f16.
// Weights pre-converted once per launch to f16 B^T [N][K]; the GEMM k-loop is
// a double-buffered async-DMA pipeline: global_load_async_to_lds_b128 for
// tile t+1 overlaps the v_wmma's on tile t (partial s_wait_asynccnt(3)).
// ---------------------------------------------------------------------------

typedef __attribute__((ext_vector_type(16))) _Float16 v16h;
typedef __attribute__((ext_vector_type(8)))  float    v8f;
typedef __attribute__((ext_vector_type(4)))  int      v4i;

#define NTOK 4096   // B*T
#define TSEQ 1024
#define MD   512
#define NH   8
#define DH   64
#define FFD  2048

// ---- CDNA5 async global->LDS copy (guarded; falls back to b128 copy) ------
#if defined(__has_builtin)
#  if __has_builtin(__builtin_amdgcn_global_load_async_to_lds_b128) && \
      __has_builtin(__builtin_amdgcn_s_wait_asynccnt)
#    define USE_ASYNC_COPY 1
#  endif
#endif
#ifndef USE_ASYNC_COPY
#  define USE_ASYNC_COPY 0
#endif

#if USE_ASYNC_COPY
typedef __attribute__((address_space(1))) v4i v4i_g;   // global
typedef __attribute__((address_space(3))) v4i v4i_l;   // LDS
#endif

__device__ __forceinline__ void copy16_to_lds(void* ldst, const void* gsrc) {
#if USE_ASYNC_COPY
  __builtin_amdgcn_global_load_async_to_lds_b128(
      (v4i_g*)(unsigned long long)gsrc,
      (v4i_l*)(unsigned)(unsigned long long)ldst,
      0, 0);
#else
  *(uint4*)ldst = *(const uint4*)gsrc;
#endif
}

// wait for ALL of this wave's async copies, then block barrier
__device__ __forceinline__ void lds_copy_fence() {
#if USE_ASYNC_COPY
  __builtin_amdgcn_s_wait_asynccnt(0);
#endif
  __syncthreads();
}

__device__ __forceinline__ v8f zero8() {
  v8f z = {0.f,0.f,0.f,0.f,0.f,0.f,0.f,0.f};
  return z;
}

// A-fragment (16x32 f16, one 32-K chunk).  base = uint ptr to this lane's row
// (K contiguous).  Layout per CDNA5 ISA 7.12.2:
//   lanes 0-15: VGPR i<4 -> K=2i,2i+1 ; VGPR 4..7 -> K=16+2(i-4),..
//   lanes16-31: VGPR i<4 -> K=8+2i    ; VGPR 4..7 -> K=24+2(i-4),..
__device__ __forceinline__ v16h load_afrag(const unsigned* base, int hi) {
  union { unsigned u[8]; v16h v; } r;
#pragma unroll
  for (int i = 0; i < 4; ++i) r.u[i]     = base[hi*4 + i];
#pragma unroll
  for (int i = 0; i < 4; ++i) r.u[4 + i] = base[8 + hi*4 + i];
  return r.v;
}

// B-fragment (32x16 f16) from LDS stored transposed [n][k].
//   same K order as A: lanes 0-15: VGPR i<4 -> K=2i,2i+1 ; VGPR 4..7 -> K=16+2(i-4),..
//   lanes16-31: VGPR i<4 -> K=8+2i ; VGPR 4..7 -> K=24+2(i-4),..
__device__ __forceinline__ v16h load_bfrag(const unsigned* colbase, int hi) {
  union { unsigned u[8]; v16h v; } r;
#pragma unroll
  for (int i = 0; i < 4; ++i) r.u[i]     = colbase[hi*4 + i];
#pragma unroll
  for (int i = 0; i < 4; ++i) r.u[4 + i] = colbase[8 + hi*4 + i];
  return r.v;
}

__device__ __forceinline__ v8f wmma32(v16h a, v16h b, v8f c) {
  return __builtin_amdgcn_wmma_f32_16x16x32_f16(
      false, a, false, b, (short)0, c, false, false);
}

// ---------------------------------------------------------------------------
// Weight prep: fp32 weight (generalized addressing) -> f16 B^T [N][K].
// src element (k, n) at  W + (n>>6)*hsB + k*ldb + (n&63).
// ---------------------------------------------------------------------------
__global__ __launch_bounds__(256) void wconv_kernel(
    const float* __restrict__ W, long hsB, int ldb, int K,
    _Float16* __restrict__ out)
{
  const int n = blockIdx.x;
  const float* src = W + (long)(n >> 6) * hsB + (n & 63);
  _Float16* dst = out + (long)n * K;
  for (int k = threadIdx.x; k < K; k += 256)
    dst[k] = (_Float16)src[(long)k * ldb];
}

// ---------------------------------------------------------------------------
// GEMM:  C[r,n] = sum_k A[r,k]*Bt[n,k] + bias[n]   (optional ReLU)
// A:  f16 row-major [Mrows x K];  Bt: f16 [N x K] (pre-transposed weights).
// C (f32 and/or f16) addressed as  C + (n>>6)*hsC + r*ldc + (n&63)  so q/k/v
// scatter straight into (H, NTOK, 64).
// Block: 128x64 tile, 256 threads = 8 waves; wave w does rows w*16..w*16+15,
// 4 WMMAs per k-step.  Double-buffered async-DMA pipeline: each wave issues 3
// async b128 copies per tile; after issuing tile t+1 it waits asynccnt<=3 so
// only tile t is guaranteed resident while t+1 streams in under the WMMAs.
// ---------------------------------------------------------------------------
__global__ __launch_bounds__(256) void gemm_f16_kernel(
    const _Float16* __restrict__ A, int K,
    const _Float16* __restrict__ Bt,
    const float* __restrict__ bias,
    float* __restrict__ Cf, _Float16* __restrict__ Ch, long hsC, int ldc,
    int relu)
{
  __shared__ __align__(16) _Float16 As[2][128 * 40];  // row stride 40 halves
  __shared__ __align__(16) _Float16 Bs[2][64 * 40];

  const int tid  = threadIdx.x;
  const int m0   = blockIdx.y * 128;
  const int n0   = blockIdx.x * 64;
  const int wave = tid >> 5, lane = tid & 31;
  const int lrow = lane & 15, hi = lane >> 4;

  const uint4* Ag4 = (const uint4*)A;
  const uint4* Bg4 = (const uint4*)Bt;
  const int strU4  = K >> 3;            // uint4 per K-row

  const int rr = tid >> 2, q = tid & 3; // staging coordinates
  const long aoff0 = (long)(m0 + rr) * strU4 + q;
  const long aoff1 = (long)(m0 + rr + 64) * strU4 + q;
  const long boff  = (long)(n0 + rr) * strU4 + q;

  v8f c[4];
#pragma unroll
  for (int ct = 0; ct < 4; ++ct) c[ct] = zero8();

  const int nk = K >> 5;                // 32-wide k tiles

  auto issue = [&](int t, int buf) {    // stage tile t into buffer buf
    uint4* As4 = (uint4*)As[buf];
    uint4* Bs4 = (uint4*)Bs[buf];
    const long ko4 = (long)t * 4;
    copy16_to_lds(&As4[rr * 5 + q],        &Ag4[aoff0 + ko4]);
    copy16_to_lds(&As4[(rr + 64) * 5 + q], &Ag4[aoff1 + ko4]);
    copy16_to_lds(&Bs4[rr * 5 + q],        &Bg4[boff + ko4]);
  };

  issue(0, 0);
  for (int t = 0; t < nk; ++t) {
    if (t + 1 < nk) {
      // overlap: DMA tile t+1 into the other buffer (WAR-safe: that buffer's
      // readers finished before the previous iteration's closing barrier)
      issue(t + 1, (t + 1) & 1);
      if (t + 2 < nk) {                 // HBM->L2 headstart for tile t+2
        __builtin_prefetch(&Ag4[aoff0 + (long)(t + 2) * 4]);
        __builtin_prefetch(&Bg4[boff  + (long)(t + 2) * 4]);
      }
#if USE_ASYNC_COPY
      __builtin_amdgcn_s_wait_asynccnt(3);  // tile t landed; t+1 in flight
#endif
    } else {
#if USE_ASYNC_COPY
      __builtin_amdgcn_s_wait_asynccnt(0);
#endif
    }
    __syncthreads();                    // tile t visible to all waves

    const unsigned* AsU = (const unsigned*)As[t & 1];  // row stride 20 uints
    const unsigned* BsU = (const unsigned*)Bs[t & 1];
    v16h a = load_afrag(AsU + (wave * 16 + lrow) * 20, hi);
#pragma unroll
    for (int ct = 0; ct < 4; ++ct)
      c[ct] = wmma32(a, load_bfrag(BsU + (ct * 16 + lrow) * 20, hi), c[ct]);
    __syncthreads();                    // all waves done reading buffer t&1
  }

  // ---- epilogue (branch-hoisted) ----
  const long cbase = (long)(n0 >> 6) * hsC;
  const float lo = relu ? 0.f : -3.0e38f;
  float bv[4];
#pragma unroll
  for (int ct = 0; ct < 4; ++ct) bv[ct] = bias[n0 + ct * 16 + lrow];

  if (Cf) {
#pragma unroll
    for (int r = 0; r < 8; ++r) {
      const int grow = m0 + wave * 16 + r + hi * 8;
#pragma unroll
      for (int ct = 0; ct < 4; ++ct) {
        float v = fmaxf(c[ct][r] + bv[ct], lo);
        Cf[cbase + (long)grow * ldc + ct * 16 + lrow] = v;
      }
    }
  }
  if (Ch) {
#pragma unroll
    for (int r = 0; r < 8; ++r) {
      const int grow = m0 + wave * 16 + r + hi * 8;
#pragma unroll
      for (int ct = 0; ct < 4; ++ct) {
        float v = fmaxf(c[ct][r] + bv[ct], lo);
        Ch[cbase + (long)grow * ldc + ct * 16 + lrow] = (_Float16)v;
      }
    }
  }
}

// ---------------------------------------------------------------------------
// Flash attention.  q/k/v: f16 (NH, NTOK, 64).  ctx: f16 (NTOK, 512) with
// per-head columns h*64..h*64+63.  grid (T/64, NH, B), 128 threads = 4 waves.
// ---------------------------------------------------------------------------
__global__ __launch_bounds__(128) void attn_kernel(
    const _Float16* __restrict__ Q, const _Float16* __restrict__ Kmat,
    const _Float16* __restrict__ Vmat, _Float16* __restrict__ ctx,
    int Tlen, int nt, int causal, float scale)
{
  __shared__ __align__(16) _Float16 Ks[64 * 72];  // [t'][d]  (B-layout for S)
  __shared__ __align__(16) _Float16 Vs[64 * 72];  // [d][t']  (B-layout for O)
  __shared__ __align__(16) _Float16 Ps[64 * 72];  // P strip per wave
  unsigned* KsU = (unsigned*)Ks;   // row stride 36 uints
  unsigned* VsU = (unsigned*)Vs;
  unsigned* PsU = (unsigned*)Ps;
  uint4*    Ks4 = (uint4*)Ks;      // row stride 9

  const int tid  = threadIdx.x;
  const int wave = tid >> 5, lane = tid & 31;
  const int lrow = lane & 15, hi = lane >> 4;
  const int h = blockIdx.y, b = blockIdx.z;
  const int q0 = blockIdx.x * 64;
  const long rowbase = (long)h * nt + (long)b * Tlen;

  // Q fragments held in registers for the whole kv loop
  const unsigned* gq =
      (const unsigned*)Q + (rowbase + q0 + wave * 16 + lrow) * 32;
  v16h aq0 = load_afrag(gq, hi);        // d = 0..31
  v16h aq1 = load_afrag(gq + 16, hi);   // d = 32..63

  float mrow[8], lsum[8];
  v8f o[4];
#pragma unroll
  for (int r = 0; r < 8; ++r) { mrow[r] = -1e30f; lsum[r] = 0.f; }
#pragma unroll
  for (int ct = 0; ct < 4; ++ct) o[ct] = zero8();

  const int nkt = causal ? (q0 >> 6) + 1 : (Tlen >> 6);
  for (int kt = 0; kt < nkt; ++kt) {
    const uint4* gk4 =
        (const uint4*)Kmat + (rowbase + kt * 64) * 8;   // 8 uint4 per row
    const unsigned* gv =
        (const unsigned*)Vmat + (rowbase + kt * 64) * 32;
    // K tile, natural [t'][d]: 512 x b128, 4 per thread (async path)
#pragma unroll
    for (int i = 0; i < 4; ++i) {
      int j = tid + i * 128;
      int r = j >> 3, qq = j & 7;
      copy16_to_lds(&Ks4[r * 9 + qq], &gk4[r * 8 + qq]);
    }
    // V tile, transposed to [d][t']
    {
      int tt = tid & 63, halfsel = tid >> 6;
#pragma unroll
      for (int j = 0; j < 16; ++j) {
        int d0 = halfsel * 32 + 2 * j;
        union { unsigned u; _Float16 h2[2]; } w;
        w.u = gv[(long)tt * 32 + (d0 >> 1)];
        Vs[d0 * 72 + tt]       = w.h2[0];
        Vs[(d0 + 1) * 72 + tt] = w.h2[1];
      }
    }
    lds_copy_fence();

    // S = Q * K^T  (16 x 64 per wave)
    v8f s[4];
#pragma unroll
    for (int ct = 0; ct < 4; ++ct) {
      s[ct] = zero8();
      const unsigned* cb = KsU + (ct * 16 + lrow) * 36;
      s[ct] = wmma32(aq0, load_bfrag(cb, hi), s[ct]);
      s[ct] = wmma32(aq1, load_bfrag(cb + 16, hi), s[ct]);
    }
    // scale + causal mask
#pragma unroll
    for (int ct = 0; ct < 4; ++ct)
#pragma unroll
      for (int r = 0; r < 8; ++r) {
        float v = s[ct][r] * scale;
        if (causal) {
          int qp = q0 + wave * 16 + r + hi * 8;
          int kp = kt * 64 + ct * 16 + lrow;
          if (kp > qp) v = -1e30f;
        }
        s[ct][r] = v;
      }
    // online softmax (rows live across the 16 lanes of a half-wave)
#pragma unroll
    for (int r = 0; r < 8; ++r) {
      float v = fmaxf(fmaxf(s[0][r], s[1][r]), fmaxf(s[2][r], s[3][r]));
      v = fmaxf(v, __shfl_xor(v, 1));
      v = fmaxf(v, __shfl_xor(v, 2));
      v = fmaxf(v, __shfl_xor(v, 4));
      v = fmaxf(v, __shfl_xor(v, 8));
      float mn = fmaxf(mrow[r], v);
      float alpha = __expf(mrow[r] - mn);
      float ps = 0.f;
#pragma unroll
      for (int ct = 0; ct < 4; ++ct) {
        float p = __expf(s[ct][r] - mn);
        s[ct][r] = p;
        ps += p;
      }
      ps += __shfl_xor(ps, 1);
      ps += __shfl_xor(ps, 2);
      ps += __shfl_xor(ps, 4);
      ps += __shfl_xor(ps, 8);
      lsum[r] = lsum[r] * alpha + ps;
#pragma unroll
      for (int ct = 0; ct < 4; ++ct) o[ct][r] *= alpha;
      mrow[r] = mn;
    }
    // convert P -> f16 via LDS (C-layout -> A-layout)
#pragma unroll
    for (int ct = 0; ct < 4; ++ct)
#pragma unroll
      for (int r = 0; r < 8; ++r)
        Ps[(wave * 16 + r + hi * 8) * 72 + ct * 16 + lrow] = (_Float16)s[ct][r];
    __syncthreads();

    const unsigned* pb = PsU + (wave * 16 + lrow) * 36;
    v16h ap0 = load_afrag(pb, hi);
    v16h ap1 = load_afrag(pb + 16, hi);
#pragma unroll
    for (int ct = 0; ct < 4; ++ct) {
      const unsigned* cb = VsU + (ct * 16 + lrow) * 36;
      o[ct] = wmma32(ap0, load_bfrag(cb, hi), o[ct]);
      o[ct] = wmma32(ap1, load_bfrag(cb + 16, hi), o[ct]);
    }
    __syncthreads();
  }

#pragma unroll
  for (int r = 0; r < 8; ++r) {
    float inv = 1.f / lsum[r];
    int grow = b * Tlen + q0 + wave * 16 + r + hi * 8;
#pragma unroll
    for (int ct = 0; ct < 4; ++ct) {
      int col = h * 64 + ct * 16 + lrow;
      ctx[(long)grow * MD + col] = (_Float16)(o[ct][r] * inv);
    }
  }
}

// ---------------------------------------------------------------------------
// residual add + LayerNorm.  one token row per block.
// ---------------------------------------------------------------------------
__global__ __launch_bounds__(256) void add_ln_kernel(
    const float* __restrict__ x, const float* __restrict__ y,
    const float* __restrict__ g, const float* __restrict__ bb,
    float* __restrict__ xo, _Float16* __restrict__ xho)
{
  __shared__ float r1[256], r2[256];
  const int row = blockIdx.x, tid = threadIdx.x;
  long base = (long)row * MD;
  float v0 = x[base + tid]       + y[base + tid];
  float v1 = x[base + tid + 256] + y[base + tid + 256];
  r1[tid] = v0 + v1;
  r2[tid] = v0 * v0 + v1 * v1;
  __syncthreads();
  for (int s = 128; s > 0; s >>= 1) {
    if (tid < s) { r1[tid] += r1[tid + s]; r2[tid] += r2[tid + s]; }
    __syncthreads();
  }
  float mu   = r1[0] * (1.f / MD);
  float var  = r2[0] * (1.f / MD) - mu * mu;
  float rstd = rsqrtf(var + 1e-5f);
  float o0 = (v0 - mu) * rstd * g[tid]       + bb[tid];
  float o1 = (v1 - mu) * rstd * g[tid + 256] + bb[tid + 256];
  xo[base + tid]        = o0;
  xo[base + tid + 256]  = o1;
  xho[base + tid]       = (_Float16)o0;
  xho[base + tid + 256] = (_Float16)o1;
}

// ---------------------------------------------------------------------------
// embedding gather + sinusoidal positional encoding
// ---------------------------------------------------------------------------
__global__ __launch_bounds__(256) void embed_kernel(
    const int* __restrict__ tok, const float* __restrict__ emb,
    float* __restrict__ xo, _Float16* __restrict__ xho, int Tlen)
{
  const int row = blockIdx.x, tid = threadIdx.x;
  const int t = row % Tlen;
  const int token = tok[row];
  long base = (long)row * MD;
#pragma unroll
  for (int i = 0; i < 2; ++i) {
    int j = tid + i * 256;
    int jj = (j < 256) ? j : (j - 256);
    float freq = __expf(-(2.f * jj * (1.f / MD)) * 9.210340371976184f); // ln 1e4
    float ph = (float)t * freq;
    float pe = (j < 256) ? sinf(ph) : cosf(ph);
    float v = emb[(long)token * MD + j] + pe;
    xo[base + j]  = v;
    xho[base + j] = (_Float16)v;
  }
}

// ---------------------------------------------------------------------------
// host side
// ---------------------------------------------------------------------------
struct AttW { const float *Wq,*bq,*Wk,*bk,*Wv,*bv,*Wo,*bo; };
struct FfnW { const float *w1,*b1,*w2,*b2; };
struct EncL { AttW att; const float *g1,*be1,*g2,*be2; FfnW f; };
struct DecL { AttW satt, catt; const float *g1,*be1,*g2,*be2,*g3,*be3; FfnW f; };
struct AttH { _Float16 *q,*k,*v,*o; };   // converted f16 B^T weights
struct FfnH { _Float16 *w1,*w2; };

static void gemm(hipStream_t s, const _Float16* A, int Mrows, int N, int K,
                 const _Float16* Bt, const float* bias,
                 float* Cf, _Float16* Ch, long hsC, int ldc, int relu)
{
  dim3 g(N / 64, Mrows / 128), blk(256);
  gemm_f16_kernel<<<g, blk, 0, s>>>(A, K, Bt, bias, Cf, Ch, hsC, ldc, relu);
}

extern "C" void kernel_launch(void* const* d_in, const int* in_sizes, int n_in,
                              void* d_out, int out_size, void* d_ws, size_t ws_size,
                              hipStream_t stream)
{
  (void)in_sizes; (void)n_in; (void)out_size; (void)ws_size;

  // ---- unpack inputs in setup_inputs() dict insertion order ----
  int idx = 0;
  auto F = [&]() { return (const float*)d_in[idx++]; };
  auto att = [&]() {
    AttW a;
    a.Wq = F(); a.bq = F(); a.Wk = F(); a.bk = F();
    a.Wv = F(); a.bv = F(); a.Wo = F(); a.bo = F();
    return a;
  };
  const float* enc_emb = F();
  const float* dec_emb = F();
  EncL enc[2]; DecL dec[2];
  for (int l = 0; l < 2; ++l) {
    enc[l].att = att();
    enc[l].g1 = F(); enc[l].be1 = F(); enc[l].g2 = F(); enc[l].be2 = F();
    enc[l].f.w1 = F(); enc[l].f.b1 = F(); enc[l].f.w2 = F(); enc[l].f.b2 = F();
  }
  for (int l = 0; l < 2; ++l) {
    dec[l].satt = att(); dec[l].catt = att();
    dec[l].g1 = F(); dec[l].be1 = F(); dec[l].g2 = F(); dec[l].be2 = F();
    dec[l].g3 = F(); dec[l].be3 = F();
    dec[l].f.w1 = F(); dec[l].f.b1 = F(); dec[l].f.w2 = F(); dec[l].f.b2 = F();
  }
  const float* out_w = F();
  const float* out_b = F();
  const int* enc_in = (const int*)d_in[idx++];
  const int* tgt    = (const int*)d_in[idx++];

  // ---- workspace carve ----
  char* ws = (char*)d_ws;
  size_t off = 0;
  auto carve = [&](size_t bytes) {
    void* p = ws + off;
    off += (bytes + 255) & ~(size_t)255;
    return p;
  };
  float*    x      = (float*)   carve((size_t)NTOK * MD * 4);
  _Float16* xh     = (_Float16*)carve((size_t)NTOK * MD * 2);
  _Float16* xh_enc = (_Float16*)carve((size_t)NTOK * MD * 2);
  float*    y      = (float*)   carve((size_t)NTOK * MD * 4);
  _Float16* qh     = (_Float16*)carve((size_t)NH * NTOK * DH * 2);
  _Float16* kh     = (_Float16*)carve((size_t)NH * NTOK * DH * 2);
  _Float16* vh     = (_Float16*)carve((size_t)NH * NTOK * DH * 2);
  _Float16* ctxh   = (_Float16*)carve((size_t)NTOK * MD * 2);
  _Float16* h1h    = (_Float16*)carve((size_t)NTOK * FFD * 2);

  // converted f16 weights (B^T [N][K])
  auto carveH = [&](size_t elems) { return (_Float16*)carve(elems * 2); };
  const long hsW_head = (long)MD * DH;   // (H, M, 64) source weights

  auto conv_att = [&](const AttW& w) {
    AttH hw;
    hw.q = carveH((size_t)MD * MD);
    hw.k = carveH((size_t)MD * MD);
    hw.v = carveH((size_t)MD * MD);
    hw.o = carveH((size_t)MD * MD);
    wconv_kernel<<<MD, 256, 0, stream>>>(w.Wq, hsW_head, DH, MD, hw.q);
    wconv_kernel<<<MD, 256, 0, stream>>>(w.Wk, hsW_head, DH, MD, hw.k);
    wconv_kernel<<<MD, 256, 0, stream>>>(w.Wv, hsW_head, DH, MD, hw.v);
    wconv_kernel<<<MD, 256, 0, stream>>>(w.Wo, 64, MD, MD, hw.o);
    return hw;
  };
  auto conv_ffn = [&](const FfnW& f) {
    FfnH hf;
    hf.w1 = carveH((size_t)MD * FFD);
    hf.w2 = carveH((size_t)FFD * MD);
    wconv_kernel<<<FFD, 256, 0, stream>>>(f.w1, 64, FFD, MD, hf.w1);
    wconv_kernel<<<MD, 256, 0, stream>>>(f.w2, 64, MD, FFD, hf.w2);
    return hf;
  };

  AttH enc_att[2], dec_satt[2], dec_catt[2];
  FfnH enc_ffn[2], dec_ffn[2];
  for (int l = 0; l < 2; ++l) { enc_att[l] = conv_att(enc[l].att); enc_ffn[l] = conv_ffn(enc[l].f); }
  for (int l = 0; l < 2; ++l) {
    dec_satt[l] = conv_att(dec[l].satt);
    dec_catt[l] = conv_att(dec[l].catt);
    dec_ffn[l]  = conv_ffn(dec[l].f);
  }
  _Float16* out_wh = carveH((size_t)64 * MD);
  wconv_kernel<<<64, 256, 0, stream>>>(out_w, 64, 64, MD, out_wh);

  const long hsC_head = (long)NTOK * DH;   // (H, NTOK, 64) q/k/v outputs

  auto run_mha = [&](const _Float16* aq, const _Float16* akv, const AttW& w,
                     const AttH& hw, int causal) {
    gemm(stream, aq,  NTOK, MD, MD, hw.q, w.bq, nullptr, qh, hsC_head, DH, 0);
    gemm(stream, akv, NTOK, MD, MD, hw.k, w.bk, nullptr, kh, hsC_head, DH, 0);
    gemm(stream, akv, NTOK, MD, MD, hw.v, w.bv, nullptr, vh, hsC_head, DH, 0);
    attn_kernel<<<dim3(TSEQ / 64, NH, 4), 128, 0, stream>>>(
        qh, kh, vh, ctxh, TSEQ, NTOK, causal, 0.125f /* 1/sqrt(64) */);
    gemm(stream, ctxh, NTOK, MD, MD, hw.o, w.bo, y, nullptr, 64, MD, 0);
  };
  auto run_ffn = [&](const _Float16* a, const FfnW& f, const FfnH& hf) {
    gemm(stream, a,   NTOK, FFD, MD,  hf.w1, f.b1, nullptr, h1h, 64, FFD, 1);
    gemm(stream, h1h, NTOK, MD,  FFD, hf.w2, f.b2, y, nullptr, 64, MD, 0);
  };
  auto ln = [&](const float* gg, const float* bbb) {
    add_ln_kernel<<<NTOK, 256, 0, stream>>>(x, y, gg, bbb, x, xh);
  };

  // ---- encoder ----
  embed_kernel<<<NTOK, 256, 0, stream>>>(enc_in, enc_emb, x, xh, TSEQ);
  for (int l = 0; l < 2; ++l) {
    run_mha(xh, xh, enc[l].att, enc_att[l], 0);
    ln(enc[l].g1, enc[l].be1);
    run_ffn(xh, enc[l].f, enc_ffn[l]);
    ln(enc[l].g2, enc[l].be2);
  }
  (void)hipMemcpyAsync(xh_enc, xh, (size_t)NTOK * MD * 2,
                       hipMemcpyDeviceToDevice, stream);

  // ---- decoder (teacher forcing) ----
  embed_kernel<<<NTOK, 256, 0, stream>>>(tgt, dec_emb, x, xh, TSEQ);
  for (int l = 0; l < 2; ++l) {
    run_mha(xh, xh, dec[l].satt, dec_satt[l], 1);        // causal self-att
    ln(dec[l].g1, dec[l].be1);
    run_mha(xh, xh_enc, dec[l].catt, dec_catt[l], 0);    // cross-att
    ln(dec[l].g2, dec[l].be2);
    run_ffn(xh, dec[l].f, dec_ffn[l]);
    ln(dec[l].g3, dec[l].be3);
  }

  // ---- output projection -> d_out (fp32, 4096 x 64) ----
  gemm(stream, xh, NTOK, 64, MD, out_wh, out_b, (float*)d_out, nullptr, 64, 64, 0);
}
